// Aligner_36885179138568
// MI455X (gfx1250) — matrix operand order repaired and checked by reading, not compiled, not run on hardware
//
#include <hip/hip_runtime.h>
#include <hip/hip_bf16.h>
#include <cstdint>
#include <cstddef>

#define B_    16
#define DE_   512
#define DD_   80
#define DDP_  96     // DD padded to multiple of 32 for WMMA K-dim
#define H_    256
#define T1_   320
#define T2_   1600
#define KE_   3
#define KD_   7
#define TEMP_ 0.0005f
#define SLOPE_ 0.3f

typedef __attribute__((ext_vector_type(16))) __bf16 v16bf;
typedef __attribute__((ext_vector_type(8)))  float  v8f;

union V16u { v16bf v; uint4 q[2]; };

// ---- WMMA fragment loaders (contiguous k = 16h + s order, wave32) ----
// A tile 16x32 bf16 (MxK): lane<16 -> row r0+lane, K = c0..c0+15
//                          lane>=16 -> row r0+lane-16, K = c0+16..c0+31
__device__ inline v16bf load_a16x32(const __bf16* __restrict__ base, int r0, int c0,
                                    int ld, int lane) {
  const int r = lane & 15;
  const __bf16* p = base + (size_t)(r0 + r) * ld + c0 + ((lane < 16) ? 0 : 16);
  V16u x;
  x.q[0] = *(const uint4*)p;
  x.q[1] = *(const uint4*)(p + 8);
  return x.v;
}

// B tile 32x16 bf16 (KxN), source stored N-major rows of length ld:
// lane<16 -> col t0+lane, K = c0..c0+15 ; lane>=16 -> col t0+lane-16, K = c0+16..c0+31
__device__ inline v16bf load_b32x16_fast(const __bf16* __restrict__ base, int t0, int c0,
                                         int ld, int lane) {
  const int n = lane & 15;
  const __bf16* p = base + (size_t)(t0 + n) * ld + c0 + ((lane < 16) ? 0 : 16);
  V16u x;
  x.q[0] = *(const uint4*)p;
  x.q[1] = *(const uint4*)(p + 8);
  return x.v;
}

// Checked variant: zero-fills out-of-range time steps (conv boundary padding).
__device__ inline v16bf load_b32x16_chk(const __bf16* __restrict__ base, int t0, int c0,
                                        int ld, int lane, int Tlim) {
  const int n = lane & 15;
  const int t = t0 + n;
  V16u x;
  if (t >= 0 && t < Tlim) {
    const __bf16* p = base + (size_t)t * ld + c0 + ((lane < 16) ? 0 : 16);
    x.q[0] = *(const uint4*)p;
    x.q[1] = *(const uint4*)(p + 8);
  } else {
    x.q[0] = make_uint4(0u, 0u, 0u, 0u);
    x.q[1] = make_uint4(0u, 0u, 0u, 0u);
  }
  return x.v;
}

// ---- layout conversion: [B,C,T] f32 -> [B,T,Cp] bf16 (zero-padded channels) ----
__global__ void transpose_pad_kernel(const float* __restrict__ in, __bf16* __restrict__ out,
                                     int C, int Cp, int T) {
  size_t idx = (size_t)blockIdx.x * blockDim.x + threadIdx.x;
  size_t total = (size_t)B_ * T * Cp;
  if (idx >= total) return;
  int c = (int)(idx % Cp);
  size_t bt = idx / Cp;
  int t = (int)(bt % T);
  int b = (int)(bt / T);
  float v = (c < C) ? in[((size_t)b * C + c) * T + t] : 0.0f;
  out[idx] = (__bf16)v;
}

// ---- weights: [H,C,K] f32 -> [K,H,Cp] bf16 (zero-padded channels) ----
__global__ void weight_cvt_kernel(const float* __restrict__ w, __bf16* __restrict__ out,
                                  int C, int Cp, int K) {
  size_t idx = (size_t)blockIdx.x * blockDim.x + threadIdx.x;
  size_t total = (size_t)K * H_ * Cp;
  if (idx >= total) return;
  int c = (int)(idx % Cp);
  int h = (int)((idx / Cp) % H_);
  int dk = (int)(idx / ((size_t)Cp * H_));
  float v = (c < C) ? w[((size_t)h * C + c) * K + dk] : 0.0f;
  out[idx] = (__bf16)v;
}

// ---- fused conv1d (as K shifted GEMMs) + bias + LeakyReLU, bf16 in/out ----
// Each wave: 16*MT(H) x 16*NT(T) output tile; A fragments feed NT WMMAs, B fragments
// feed MT WMMAs. Register double-buffering: K-step kk+1 loads issue before kk's WMMAs.
template <int CIN, int KSZ, int MT, int NT, bool CHECKED>
__device__ inline void conv_body(const __bf16* __restrict__ Xb,
                                 const __bf16* __restrict__ W,
                                 const float* __restrict__ bias,
                                 __bf16* __restrict__ Y,
                                 int T, int b, int t0, int h0, int lane) {
  constexpr int KC = CIN / 32;
  constexpr int NK = KSZ * KC;
  v8f zero = {};
  v8f acc[MT][NT];
#pragma unroll
  for (int mt = 0; mt < MT; ++mt)
#pragma unroll
    for (int nt = 0; nt < NT; ++nt) acc[mt][nt] = zero;

  v16bf aB[2][MT];
  v16bf bB[2][NT];

  auto loadK = [&](int kk, int buf) {
    const int dk = kk / KC;
    const int c0 = (kk % KC) * 32;
    const __bf16* Wd = W + (size_t)dk * H_ * CIN;
    const int shift = dk - KSZ / 2;
#pragma unroll
    for (int mt = 0; mt < MT; ++mt)
      aB[buf][mt] = load_a16x32(Wd, h0 + mt * 16, c0, CIN, lane);
#pragma unroll
    for (int nt = 0; nt < NT; ++nt) {
      if (CHECKED)
        bB[buf][nt] = load_b32x16_chk(Xb, t0 + nt * 16 + shift, c0, CIN, lane, T);
      else
        bB[buf][nt] = load_b32x16_fast(Xb, t0 + nt * 16 + shift, c0, CIN, lane);
    }
  };

  loadK(0, 0);
#pragma unroll
  for (int kk = 0; kk < NK; ++kk) {
    const int cur = kk & 1;
    if (kk + 1 < NK) loadK(kk + 1, cur ^ 1);
#pragma unroll
    for (int mt = 0; mt < MT; ++mt)
#pragma unroll
      for (int nt = 0; nt < NT; ++nt)
        acc[mt][nt] = __builtin_amdgcn_wmma_f32_16x16x32_bf16(
            false, aB[cur][mt], false, bB[cur][nt], (short)0, acc[mt][nt], false, false);
  }

  // D layout: elem m -> row +m (+8 for lanes 16..31), col = strip col + (lane&15)
  const int n = lane & 15;
#pragma unroll
  for (int mt = 0; mt < MT; ++mt) {
    const int hb = h0 + mt * 16 + ((lane < 16) ? 0 : 8);
    float bv[8];
#pragma unroll
    for (int m = 0; m < 8; ++m) bv[m] = bias[hb + m];
#pragma unroll
    for (int nt = 0; nt < NT; ++nt) {
      const int t = t0 + nt * 16 + n;
      __bf16 tmp[8];
#pragma unroll
      for (int m = 0; m < 8; ++m) {
        float v = acc[mt][nt][m] + bv[m];
        v = (v >= 0.0f) ? v : SLOPE_ * v;
        tmp[m] = (__bf16)v;
      }
      *(uint4*)(Y + ((size_t)b * T + t) * H_ + hb) = *(const uint4*)tmp;
    }
  }
}

template <int CIN, int KSZ, int MT, int NT>
__global__ __launch_bounds__(128, 1)
void conv_wmma_kernel(const __bf16* __restrict__ X,  // [B,T,CIN]
                      const __bf16* __restrict__ W,  // [KSZ,H,CIN]
                      const float* __restrict__ bias,
                      __bf16* __restrict__ Y,        // [B,T,H]
                      int T) {
  const int lane = threadIdx.x & 31;
  const int wave = threadIdx.x >> 5;
  const int t0 = blockIdx.x * (16 * NT);
  const int h0 = blockIdx.y * (4 * 16 * MT) + wave * (16 * MT);
  const int b = blockIdx.z;
  const __bf16* Xb = X + (size_t)b * T * CIN;
  // Block-uniform dispatch: only edge blocks pay for boundary predication.
  const bool interior = (t0 - KSZ / 2 >= 0) && (t0 + 16 * NT - 1 + KSZ / 2 < T);
  if (interior)
    conv_body<CIN, KSZ, MT, NT, false>(Xb, W, bias, Y, T, b, t0, h0, lane);
  else
    conv_body<CIN, KSZ, MT, NT, true>(Xb, W, bias, Y, T, b, t0, h0, lane);
}

// ---- qk = q^T k per batch via WMMA: q [B,T2,H], k [B,T1,H] -> out [B,T2,T1] f32 ----
// Each wave: 32(t2) x 64(t1) tile (2x4 accumulators); 2 waves/block; exact tiling.
__global__ __launch_bounds__(64, 1)
void qk_wmma_kernel(const __bf16* __restrict__ q,
                    const __bf16* __restrict__ k,
                    float* __restrict__ out) {
  const int lane = threadIdx.x & 31;
  const int wave = threadIdx.x >> 5;
  const int t10 = blockIdx.x * 64;
  const int t20 = (blockIdx.y * 2 + wave) * 32;
  const int b = blockIdx.z;
  const __bf16* qb = q + (size_t)b * T2_ * H_;
  const __bf16* kb = k + (size_t)b * T1_ * H_;
  constexpr int NK = H_ / 32;
  v8f zero = {};
  v8f acc[2][4];
#pragma unroll
  for (int mt = 0; mt < 2; ++mt)
#pragma unroll
    for (int nt = 0; nt < 4; ++nt) acc[mt][nt] = zero;

  v16bf aB[2][2];
  v16bf bB[2][4];

  auto loadK = [&](int kk, int buf) {
    const int c0 = kk * 32;
#pragma unroll
    for (int mt = 0; mt < 2; ++mt)
      aB[buf][mt] = load_a16x32(qb, t20 + mt * 16, c0, H_, lane);
#pragma unroll
    for (int nt = 0; nt < 4; ++nt)
      bB[buf][nt] = load_b32x16_fast(kb, t10 + nt * 16, c0, H_, lane);
  };

  loadK(0, 0);
#pragma unroll
  for (int kk = 0; kk < NK; ++kk) {
    const int cur = kk & 1;
    if (kk + 1 < NK) loadK(kk + 1, cur ^ 1);
#pragma unroll
    for (int mt = 0; mt < 2; ++mt)
#pragma unroll
      for (int nt = 0; nt < 4; ++nt)
        acc[mt][nt] = __builtin_amdgcn_wmma_f32_16x16x32_bf16(
            false, aB[cur][mt], false, bB[cur][nt], (short)0, acc[mt][nt], false, false);
  }

  const int n = lane & 15;
  const int moff = (lane < 16) ? 0 : 8;
#pragma unroll
  for (int mt = 0; mt < 2; ++mt)
#pragma unroll
    for (int nt = 0; nt < 4; ++nt)
#pragma unroll
      for (int m = 0; m < 8; ++m)
        out[((size_t)b * T2_ + (t20 + mt * 16 + moff + m)) * T1_ + t10 + nt * 16 + n] =
            acc[mt][nt][m];
}

// ---- |k|^2 per (b,t1) from bf16 [B,T1,H] ----
__global__ void ksq_kernel(const __bf16* __restrict__ k, float* __restrict__ ksq) {
  int i = blockIdx.x * blockDim.x + threadIdx.x;
  if (i >= B_ * T1_) return;
  const __bf16* p = k + (size_t)i * H_;
  float s = 0.0f;
  for (int c = 0; c < H_; ++c) {
    float v = (float)p[c];
    s += v * v;
  }
  ksq[i] = s;
}

__device__ inline float wave_max(float v) {
#pragma unroll
  for (int o = 16; o > 0; o >>= 1) v = fmaxf(v, __shfl_xor(v, o, 32));
  return v;
}
__device__ inline float wave_sum(float v) {
#pragma unroll
  for (int o = 16; o > 0; o >>= 1) v += __shfl_xor(v, o, 32);
  return v;
}

// ---- per-row (b,t2): logprob = logsoftmax(-TEMP*(ksq-2qk)) + log(prior^T+1e-8)
//      soft = softmax(logprob); masin = log(soft + 1e-12). One wave per row.
//      NOTE: q_sq is constant along t1 so it cancels in log_softmax.
__global__ void softmax_kernel(float* __restrict__ inout,        // raw qk -> logprob
                               const float* __restrict__ ksq,    // [B,T1]
                               const float* __restrict__ prior,  // [B,T1,T2]
                               float* __restrict__ soft,
                               float* __restrict__ masin) {
  const int lane = threadIdx.x & 31;
  const int row = blockIdx.x * (blockDim.x >> 5) + (threadIdx.x >> 5);
  if (row >= B_ * T2_) return;
  const int b = row / T2_;
  const int t2 = row % T2_;
  float* r = inout + (size_t)row * T1_;
  float vals[T1_ / 32];
  float m = -1e30f;
#pragma unroll
  for (int i = 0; i < T1_ / 32; ++i) {
    int j = lane + 32 * i;
    float s = -TEMP_ * (ksq[b * T1_ + j] - 2.0f * r[j]);
    vals[i] = s;
    m = fmaxf(m, s);
  }
  m = wave_max(m);
  float sum = 0.0f;
#pragma unroll
  for (int i = 0; i < T1_ / 32; ++i) sum += __expf(vals[i] - m);
  sum = wave_sum(sum);
  const float lse = m + __logf(sum);
  float m2 = -1e30f;
#pragma unroll
  for (int i = 0; i < T1_ / 32; ++i) {
    int j = lane + 32 * i;
    float lp = vals[i] - lse + __logf(prior[((size_t)b * T1_ + j) * T2_ + t2] + 1e-8f);
    vals[i] = lp;
    m2 = fmaxf(m2, lp);
  }
  m2 = wave_max(m2);
  float s2 = 0.0f;
#pragma unroll
  for (int i = 0; i < T1_ / 32; ++i) s2 += __expf(vals[i] - m2);
  s2 = wave_sum(s2);
  const float l2 = m2 + __logf(s2);
  float* so = soft + (size_t)row * T1_;
  float* mi = masin + (size_t)row * T1_;
#pragma unroll
  for (int i = 0; i < T1_ / 32; ++i) {
    int j = lane + 32 * i;
    float lp = vals[i];
    r[j] = lp;
    float sm = __expf(lp - l2);
    so[j] = sm;
    mi[j] = __logf(sm + 1e-12f);
  }
}

// ---- Monotonic alignment search: forward DP in LDS, then single-thread backtrack ----
__global__ void mas_kernel(const float* __restrict__ logp,      // [B,T2,T1]
                           unsigned char* __restrict__ choices, // [B,T2-1,T1]
                           int* __restrict__ path) {            // [B,T2]
  const int b = blockIdx.x;
  const int j = threadIdx.x;  // 0..T1-1
  __shared__ float prev[T1_];
  const float NEG = -1e9f;
  prev[j] = (j == 0) ? logp[(size_t)b * T2_ * T1_ + j] : NEG;
  __syncthreads();
  for (int t = 1; t < T2_; ++t) {
    float s = prev[j];
    float sl = (j > 0) ? prev[j - 1] : NEG;
    bool came = sl > s;  // tie -> stay (matches shifted > prev)
    float dp = logp[((size_t)b * T2_ + t) * T1_ + j] + (came ? sl : s);
    choices[((size_t)b * (T2_ - 1) + (t - 1)) * T1_ + j] = (unsigned char)came;
    __syncthreads();
    prev[j] = dp;
    __syncthreads();
  }
  if (j == 0) {
    int cur = T1_ - 1;
    path[b * T2_ + T2_ - 1] = cur;
    for (int t = T2_ - 2; t >= 0; --t) {
      cur -= (int)choices[((size_t)b * (T2_ - 1) + t) * T1_ + cur];
      path[b * T2_ + t] = cur;
    }
  }
}

__global__ void hard_fill_kernel(const int* __restrict__ path, float* __restrict__ hard) {
  int row = blockIdx.x * blockDim.x + threadIdx.x;  // b*T2 + t2
  if (row >= B_ * T2_) return;
  int pj = path[row];
  float* o = hard + (size_t)row * T1_;
  for (int jj = 0; jj < T1_; ++jj) o[jj] = (jj == pj) ? 1.0f : 0.0f;
}

__global__ void dur_kernel(const int* __restrict__ path, float* __restrict__ dur) {
  int i = blockIdx.x * blockDim.x + threadIdx.x;  // b*T1 + j
  if (i >= B_ * T1_) return;
  int b = i / T1_;
  int jj = i % T1_;
  int cnt = 0;
  for (int t = 0; t < T2_; ++t) cnt += (path[b * T2_ + t] == jj) ? 1 : 0;
  dur[i] = (float)cnt;
}

extern "C" void kernel_launch(void* const* d_in, const int* in_sizes, int n_in,
                              void* d_out, int out_size, void* d_ws, size_t ws_size,
                              hipStream_t stream) {
  const float* enc_in = (const float*)d_in[0];
  const float* dec_in = (const float*)d_in[1];
  const float* attn_prior = (const float*)d_in[5];
  const float* k1_w = (const float*)d_in[6];
  const float* k1_b = (const float*)d_in[7];
  const float* k2_w = (const float*)d_in[8];
  const float* k2_b = (const float*)d_in[9];
  const float* q1_w = (const float*)d_in[10];
  const float* q1_b = (const float*)d_in[11];
  const float* q2_w = (const float*)d_in[12];
  const float* q2_b = (const float*)d_in[13];
  const float* q3_w = (const float*)d_in[14];
  const float* q3_b = (const float*)d_in[15];

  const size_t N3 = (size_t)B_ * T2_ * T1_;
  float* out_logprob = (float*)d_out;       // reused as raw-qk scratch first
  float* out_soft = out_logprob + N3;
  float* out_hard = out_soft + N3;          // reused as MAS-input scratch first
  float* out_dur = out_hard + N3;

  // workspace carve-up (256B aligned)
  char* wsp = (char*)d_ws;
  auto alloc = [&](size_t bytes) -> void* {
    void* p = (void*)wsp;
    wsp += (bytes + 255) & ~(size_t)255;
    return p;
  };
  __bf16* Xe  = (__bf16*)alloc((size_t)B_ * T1_ * DE_ * 2);
  __bf16* Xd  = (__bf16*)alloc((size_t)B_ * T2_ * DDP_ * 2);
  __bf16* Wk1 = (__bf16*)alloc((size_t)KE_ * H_ * DE_ * 2);
  __bf16* Wk2 = (__bf16*)alloc((size_t)KE_ * H_ * H_ * 2);
  __bf16* Wq1 = (__bf16*)alloc((size_t)KD_ * H_ * DDP_ * 2);
  __bf16* Wq2 = (__bf16*)alloc((size_t)KD_ * H_ * H_ * 2);
  __bf16* Wq3 = (__bf16*)alloc((size_t)KD_ * H_ * H_ * 2);
  __bf16* kA  = (__bf16*)alloc((size_t)B_ * T1_ * H_ * 2);
  __bf16* kB  = (__bf16*)alloc((size_t)B_ * T1_ * H_ * 2);
  __bf16* qA  = (__bf16*)alloc((size_t)B_ * T2_ * H_ * 2);
  __bf16* qB  = (__bf16*)alloc((size_t)B_ * T2_ * H_ * 2);
  float* ksq  = (float*)alloc((size_t)B_ * T1_ * 4);
  unsigned char* choices = (unsigned char*)alloc((size_t)B_ * (T2_ - 1) * T1_);
  int* path   = (int*)alloc((size_t)B_ * T2_ * 4);

  // 1) layout / precision conversion
  {
    size_t n = (size_t)B_ * T1_ * DE_;
    transpose_pad_kernel<<<(unsigned)((n + 255) / 256), 256, 0, stream>>>(enc_in, Xe, DE_, DE_, T1_);
    n = (size_t)B_ * T2_ * DDP_;
    transpose_pad_kernel<<<(unsigned)((n + 255) / 256), 256, 0, stream>>>(dec_in, Xd, DD_, DDP_, T2_);
    n = (size_t)KE_ * H_ * DE_;
    weight_cvt_kernel<<<(unsigned)((n + 255) / 256), 256, 0, stream>>>(k1_w, Wk1, DE_, DE_, KE_);
    n = (size_t)KE_ * H_ * H_;
    weight_cvt_kernel<<<(unsigned)((n + 255) / 256), 256, 0, stream>>>(k2_w, Wk2, H_, H_, KE_);
    n = (size_t)KD_ * H_ * DDP_;
    weight_cvt_kernel<<<(unsigned)((n + 255) / 256), 256, 0, stream>>>(q1_w, Wq1, DD_, DDP_, KD_);
    n = (size_t)KD_ * H_ * H_;
    weight_cvt_kernel<<<(unsigned)((n + 255) / 256), 256, 0, stream>>>(q2_w, Wq2, H_, H_, KD_);
    weight_cvt_kernel<<<(unsigned)((n + 255) / 256), 256, 0, stream>>>(q3_w, Wq3, H_, H_, KD_);
  }

  // 2) conv stacks via WMMA (block: 128 H x 64 T; 4 waves; 2x4 tiles per wave)
  conv_wmma_kernel<DE_, KE_, 2, 4><<<dim3(T1_ / 64, H_ / 128, B_), 128, 0, stream>>>(Xe, Wk1, k1_b, kA, T1_);
  conv_wmma_kernel<H_, KE_, 2, 4><<<dim3(T1_ / 64, H_ / 128, B_), 128, 0, stream>>>(kA, Wk2, k2_b, kB, T1_);
  conv_wmma_kernel<DDP_, KD_, 2, 4><<<dim3(T2_ / 64, H_ / 128, B_), 128, 0, stream>>>(Xd, Wq1, q1_b, qA, T2_);
  conv_wmma_kernel<H_, KD_, 2, 4><<<dim3(T2_ / 64, H_ / 128, B_), 128, 0, stream>>>(qA, Wq2, q2_b, qB, T2_);
  conv_wmma_kernel<H_, KD_, 2, 4><<<dim3(T2_ / 64, H_ / 128, B_), 128, 0, stream>>>(qB, Wq3, q3_b, qA, T2_);

  // 3) |k|^2 and qk GEMM (raw qk staged into the logprob output region)
  ksq_kernel<<<(B_ * T1_ + 255) / 256, 256, 0, stream>>>(kB, ksq);
  qk_wmma_kernel<<<dim3(T1_ / 64, T2_ / 64, B_), 64, 0, stream>>>(qA, kB, out_logprob);

  // 4) row-wise log_softmax + prior + softmax (one wave / row); MAS input staged in out_hard
  softmax_kernel<<<(B_ * T2_) / 8, 256, 0, stream>>>(out_logprob, ksq, attn_prior,
                                                     out_soft, out_hard);

  // 5) monotonic alignment search + hard one-hot + durations
  mas_kernel<<<B_, T1_, 0, stream>>>(out_hard, choices, path);
  hard_fill_kernel<<<(B_ * T2_ + 255) / 256, 256, 0, stream>>>(path, out_hard);
  dur_kernel<<<(B_ * T1_ + 255) / 256, 256, 0, stream>>>(path, out_dur);
}
